// GatedAttentionUnit_56796647522742
// MI455X (gfx1250) — compile-verified, matrix operand order checked by reading, not run on hardware
//
#include <hip/hip_runtime.h>
#include <hip/hip_bf16.h>

// ---------------------------------------------------------------------------
// Gated Attention Unit (linear attention), MI455X / gfx1250, wave32 + WMMA.
// All heavy math runs through v_wmma_f32_16x16x32_bf16 with f32 accumulation.
// Workgroup tile 128x128 (8 waves, 4x2), wave tile 32x64 = 2x4 WMMA frags.
// kv-stage uses split-K + global_atomic_add_f32 for occupancy.
// ---------------------------------------------------------------------------

#define NB_   4          // batch
#define TQ_   4096
#define TK_   4096
#define DM_   1024       // d_model
#define SD_   512        // qk_dim
#define EPSV  1e-6f

typedef __attribute__((ext_vector_type(16))) __bf16 v16bf;
typedef __attribute__((ext_vector_type(8)))  float  v8f;

enum { EPI_NONE = 0, EPI_SILU = 1, EPI_RELU2 = 2, EPI_GAU = 3 };

__device__ __forceinline__ unsigned short f2bf(float x) {
  unsigned int u = __float_as_uint(x);
  u += 0x7fffu + ((u >> 16) & 1u);          // round-to-nearest-even
  return (unsigned short)(u >> 16);
}
__device__ __forceinline__ float bf2f(unsigned short h) {
  return __uint_as_float(((unsigned int)h) << 16);
}
// pack two f32 -> two bf16 in one dword (HW packed convert when available)
__device__ __forceinline__ unsigned int pack2_bf16(float x, float y) {
#if __has_builtin(__builtin_amdgcn_cvt_pk_bf16_f32)
  typedef __attribute__((ext_vector_type(2))) __bf16 v2bf;
  v2bf p = __builtin_amdgcn_cvt_pk_bf16_f32(x, y);
  return __builtin_bit_cast(unsigned int, p);
#else
  return (unsigned)f2bf(x) | ((unsigned)f2bf(y) << 16);
#endif
}

union FragU { v16bf v; uint4 q[2]; };

__device__ __forceinline__ v8f wmma_bf16(v16bf a, v16bf b, v8f c) {
  // (neg_a, A, neg_b, B, c_mod, C, reuse_a, reuse_b)
  return __builtin_amdgcn_wmma_f32_16x16x32_bf16(false, a, false, b, (short)0, c,
                                                 false, false);
}

// Tiling: 256 threads = 8 wave32 waves (4 along M, 2 along N).
// Workgroup tile 128(M) x 128(N), K-step 32. Wave tile 32x64 = 2x4 WMMA frags.
constexpr int BM = 128, BN = 128, BK = 32, NT = 256, LP = 40; // LP: lds pitch, mult of 8 -> 16B-aligned frag reads

// C[m,n] = epi( sum_k A[m,k] * Bw[n,k] + bias[n] )
// A:  f32 or bf16; A_T -> A[m,k] stored at A[k*lda + m]
// Bw: f32 or bf16; B_T -> Bw[n,k] stored at B[k*ldb + n]
// ATOMIC: f32 atomic-add epilogue (split-K); blockIdx.z = batch*ksplit + seg
template<bool A_F32, bool A_T, bool B_F32, bool B_T, bool OUT_BF16, int EPI,
         bool ROWMASK, bool ATOMIC>
__global__ __launch_bounds__(NT)
void gau_gemm(const void* __restrict__ Ap, long lda, long bsA,
              const void* __restrict__ Bp, long ldb, long bsB,
              const float* __restrict__ bias,
              void* __restrict__ Cp, long ldc, long bsC,
              const unsigned char* __restrict__ rowmask, long maskStride,
              const float* __restrict__ zvec,
              const float* __restrict__ umat, long bsU,
              int M, int N, int K, int ksplit)
{
  __shared__ __align__(16) unsigned short lds_a[BM * LP];
  __shared__ __align__(16) unsigned short lds_b[BN * LP];

  const int  tid = threadIdx.x;
  int batch, klo, khi;
  if constexpr (ATOMIC) {
    batch = blockIdx.z / ksplit;
    const int seg = blockIdx.z % ksplit;
    const int kc  = K / ksplit;
    klo = seg * kc; khi = klo + kc;
  } else {
    batch = blockIdx.z; klo = 0; khi = K;
  }
  const long m0 = (long)blockIdx.y * BM;
  const long n0 = (long)blockIdx.x * BN;

  const float*          A32 = nullptr; const unsigned short* A16 = nullptr;
  if constexpr (A_F32) A32 = (const float*)Ap + (long)batch * bsA;
  else                 A16 = (const unsigned short*)Ap + (long)batch * bsA;
  const float*          B32 = nullptr; const unsigned short* B16 = nullptr;
  if constexpr (B_F32) B32 = (const float*)Bp + (long)batch * bsB;
  else                 B16 = (const unsigned short*)Bp + (long)batch * bsB;

  const int wave = tid >> 5, lane = tid & 31;
  const int wm = wave & 3, wn = wave >> 2;      // 4 x 2 wave grid
  const int hl = lane >> 4, r = lane & 15;      // half-wave, row-in-half

  v8f acc[2][4] = {};

  for (long k0 = klo; k0 < khi; k0 += BK) {
    // ---------------- stage A tile (BM x BK) into LDS as bf16 ----------------
    if constexpr (!A_T) {
      #pragma unroll
      for (int i = 0; i < (BM * BK) / (NT * 4); ++i) {
        int v = tid + NT * i;
        int row = v >> 3;                 // 8 vec4 per 32-wide row
        int col = (v & 7) << 2;
        long g = (m0 + row) * lda + (k0 + col);
        if constexpr (A_F32) {
          float4 f = *(const float4*)(A32 + g);
          if (k0 + BK < khi) __builtin_prefetch((const void*)(A32 + g + BK), 0, 1);
          uint2 t; t.x = pack2_bf16(f.x, f.y); t.y = pack2_bf16(f.z, f.w);
          *(uint2*)&lds_a[row * LP + col] = t;
        } else {
          if (k0 + BK < khi) __builtin_prefetch((const void*)(A16 + g + BK), 0, 1);
          *(uint2*)&lds_a[row * LP + col] = *(const uint2*)(A16 + g);
        }
      }
    } else {
      // transposed A (bf16 source): element (m,k) at A16[(k0+k)*lda + m0+m]
      #pragma unroll
      for (int i = 0; i < (BM * BK) / (NT * 4); ++i) {
        int v  = tid + NT * i;
        int k  = v >> 5;                  // BM/4 == 32 vec4 per k row
        int mq = (v & 31) << 2;
        long g = (k0 + k) * lda + (m0 + mq);
        uint2 d = *(const uint2*)(A16 + g);
        lds_a[(mq + 0) * LP + k] = (unsigned short)(d.x);
        lds_a[(mq + 1) * LP + k] = (unsigned short)(d.x >> 16);
        lds_a[(mq + 2) * LP + k] = (unsigned short)(d.y);
        lds_a[(mq + 3) * LP + k] = (unsigned short)(d.y >> 16);
      }
    }

    // ---------------- stage B tile (BN x BK) into LDS as bf16 ----------------
    if constexpr (!B_T) {
      #pragma unroll
      for (int i = 0; i < (BN * BK) / (NT * 4); ++i) {
        int v = tid + NT * i;
        int row = v >> 3;
        int col = (v & 7) << 2;
        long g = (n0 + row) * ldb + (k0 + col);
        if constexpr (B_F32) {
          float4 f = *(const float4*)(B32 + g);
          uint2 t; t.x = pack2_bf16(f.x, f.y); t.y = pack2_bf16(f.z, f.w);
          *(uint2*)&lds_b[row * LP + col] = t;
        } else {
          *(uint2*)&lds_b[row * LP + col] = *(const uint2*)(B16 + g);
        }
      }
    } else {
      #pragma unroll
      for (int i = 0; i < (BN * BK) / (NT * 4); ++i) {
        int v  = tid + NT * i;
        int k  = v >> 5;                  // BN/4 == 32 vec4 per k row
        int nq = (v & 31) << 2;
        long g = (k0 + k) * ldb + (n0 + nq);
        if constexpr (B_F32) {
          float4 f = *(const float4*)(B32 + g);
          lds_b[(nq + 0) * LP + k] = f2bf(f.x);
          lds_b[(nq + 1) * LP + k] = f2bf(f.y);
          lds_b[(nq + 2) * LP + k] = f2bf(f.z);
          lds_b[(nq + 3) * LP + k] = f2bf(f.w);
        } else {
          uint2 d = *(const uint2*)(B16 + g);
          lds_b[(nq + 0) * LP + k] = (unsigned short)(d.x);
          lds_b[(nq + 1) * LP + k] = (unsigned short)(d.x >> 16);
          lds_b[(nq + 2) * LP + k] = (unsigned short)(d.y);
          lds_b[(nq + 3) * LP + k] = (unsigned short)(d.y >> 16);
        }
      }
    }

    __syncthreads();

    // -------- fragment loads per CDNA5 16-bit A/B layout: two b128 each -----
    FragU a[2], b[4];
    #pragma unroll
    for (int mt = 0; mt < 2; ++mt) {
      const unsigned short* pa = &lds_a[(wm * 32 + mt * 16 + r) * LP];
      a[mt].q[0] = *(const uint4*)(pa + hl * 8);
      a[mt].q[1] = *(const uint4*)(pa + 16 + hl * 8);
    }
    #pragma unroll
    for (int nt = 0; nt < 4; ++nt) {
      const unsigned short* pb = &lds_b[(wn * 64 + nt * 16 + r) * LP];
      b[nt].q[0] = *(const uint4*)(pb + hl * 8);
      b[nt].q[1] = *(const uint4*)(pb + 16 + hl * 8);
    }

    #pragma unroll
    for (int mt = 0; mt < 2; ++mt)
      #pragma unroll
      for (int nt = 0; nt < 4; ++nt)
        acc[mt][nt] = wmma_bf16(a[mt].v, b[nt].v, acc[mt][nt]);

    __syncthreads();
  }

  // ----------------------------- fused epilogue -----------------------------
  #pragma unroll
  for (int mt = 0; mt < 2; ++mt)
    #pragma unroll
    for (int nt = 0; nt < 4; ++nt)
      #pragma unroll
      for (int j = 0; j < 8; ++j) {
        long row = m0 + wm * 32 + mt * 16 + (j + 8 * hl);
        long col = n0 + wn * 64 + nt * 16 + r;
        float val = acc[mt][nt][j];
        if (bias) val += bias[col];
        if constexpr (EPI == EPI_SILU) {
          val = val / (1.f + __expf(-val));                     // x * sigmoid(x)
        } else if constexpr (EPI == EPI_RELU2) {
          float t = fmaxf(val, 0.f); val = t * t;
        } else if constexpr (EPI == EPI_GAU) {
          val = val / zvec[(long)batch * M + row];              // a = qkv / z
          val *= umat[(long)batch * bsU + row * (long)N + col]; // u * a
        }
        if constexpr (ROWMASK) val *= (float)rowmask[(long)batch * maskStride + row];
        long ci = (long)batch * bsC + row * ldc + col;
        if constexpr (ATOMIC) {
          unsafeAtomicAdd(&((float*)Cp)[ci], val);              // global_atomic_add_f32
        } else if constexpr (OUT_BF16) {
          ((unsigned short*)Cp)[ci] = f2bf(val);
        } else {
          ((float*)Cp)[ci] = val;
        }
      }
}

// k_sum[b][s] = sum_t kf[b][t][s]   (kf already key-masked)
__global__ __launch_bounds__(256)
void ksum_kernel(const unsigned short* __restrict__ kf, float* __restrict__ ksum)
{
  int b = blockIdx.y;
  int s = blockIdx.x * 256 + threadIdx.x;
  const unsigned short* p = kf + (long)b * TK_ * SD_ + s;
  float acc = 0.f;
  for (int t = 0; t < TK_; ++t) acc += bf2f(p[(long)t * SD_]);
  ksum[(long)b * SD_ + s] = acc;
}

// z[row] = dot(qf[row,:], ksum[b,:]) + eps    (one wave per row)
__global__ __launch_bounds__(256)
void z_kernel(const unsigned short* __restrict__ qf, const float* __restrict__ ksum,
              float* __restrict__ z)
{
  int wave = threadIdx.x >> 5, lane = threadIdx.x & 31;
  long row = (long)blockIdx.x * 8 + wave;
  int  b   = (int)(row / TQ_);
  const unsigned short* p  = qf + row * SD_;
  const float*          ks = ksum + (long)b * SD_;
  float acc = 0.f;
  for (int i = lane; i < SD_; i += 32) acc += bf2f(p[i]) * ks[i];
  #pragma unroll
  for (int off = 16; off > 0; off >>= 1) acc += __shfl_xor(acc, off, 32);
  if (lane == 0) z[row] = acc + EPSV;
}

extern "C" void kernel_launch(void* const* d_in, const int* in_sizes, int n_in,
                              void* d_out, int out_size, void* d_ws, size_t ws_size,
                              hipStream_t stream)
{
  (void)in_sizes; (void)n_in; (void)out_size; (void)ws_size;
  const float*         queries = (const float*)d_in[0];
  const float*         keys    = (const float*)d_in[1];
  const float*         values  = (const float*)d_in[2];
  const unsigned char* qmask   = (const unsigned char*)d_in[3];
  const unsigned char* kmask   = (const unsigned char*)d_in[4];
  const float* Wg = (const float*)d_in[5],  *bg = (const float*)d_in[6];
  const float* Wv = (const float*)d_in[7],  *bv = (const float*)d_in[8];
  const float* Wq = (const float*)d_in[9],  *bq = (const float*)d_in[10];
  const float* Wk = (const float*)d_in[11], *bk = (const float*)d_in[12];
  const float* Wo = (const float*)d_in[13], *bo = (const float*)d_in[14];

  // workspace carve-out (~168 MB)
  size_t off = 0;
  auto carve = [&](size_t bytes) {
    void* p = (char*)d_ws + off;
    off = (off + bytes + 255) & ~(size_t)255;
    return p;
  };
  float*          u     = (float*)         carve((size_t)NB_*TQ_*DM_*4); // gate, f32
  unsigned short* vfeat = (unsigned short*)carve((size_t)NB_*TK_*DM_*2); // v, bf16
  unsigned short* qfeat = (unsigned short*)carve((size_t)NB_*TQ_*SD_*2); // relu^2 q, bf16
  unsigned short* kfeat = (unsigned short*)carve((size_t)NB_*TK_*SD_*2); // relu^2 k, bf16
  float*          kv    = (float*)         carve((size_t)NB_*SD_*DM_*4); // k^T v, f32
  float*          ksum  = (float*)         carve((size_t)NB_*SD_*4);
  float*          zbuf  = (float*)         carve((size_t)NB_*TQ_*4);
  unsigned short* ua    = (unsigned short*)carve((size_t)NB_*TQ_*DM_*2); // u*a, bf16

  const dim3 blk(NT);
  const int MT = NB_ * TQ_;   // 16384 flattened rows
  const int KSPLIT = 4;

  // 1) u = SiLU(Q Wg^T + bg)                          f32 out
  gau_gemm<true,false,true,false,false,EPI_SILU,false,false>
    <<<dim3(DM_/BN, MT/BM, 1), blk, 0, stream>>>(
      queries, DM_, 0, Wg, DM_, 0, bg, u, DM_, 0,
      nullptr, 0, nullptr, nullptr, 0, MT, DM_, DM_, 1);

  // 2) v = (V Wv^T + bv) * key_mask                   bf16 out
  gau_gemm<true,false,true,false,true,EPI_NONE,true,false>
    <<<dim3(DM_/BN, MT/BM, 1), blk, 0, stream>>>(
      values, DM_, 0, Wv, DM_, 0, bv, vfeat, DM_, 0,
      kmask, 0, nullptr, nullptr, 0, MT, DM_, DM_, 1);

  // 3) qf = relu(Q Wq^T + bq)^2                       bf16 out
  gau_gemm<true,false,true,false,true,EPI_RELU2,false,false>
    <<<dim3(SD_/BN, MT/BM, 1), blk, 0, stream>>>(
      queries, DM_, 0, Wq, DM_, 0, bq, qfeat, SD_, 0,
      nullptr, 0, nullptr, nullptr, 0, MT, SD_, DM_, 1);

  // 4) kf = relu(K Wk^T + bk)^2 * key_mask            bf16 out
  gau_gemm<true,false,true,false,true,EPI_RELU2,true,false>
    <<<dim3(SD_/BN, MT/BM, 1), blk, 0, stream>>>(
      keys, DM_, 0, Wk, DM_, 0, bk, kfeat, SD_, 0,
      kmask, 0, nullptr, nullptr, 0, MT, SD_, DM_, 1);

  // 5) k_sum = sum_t kf ; 6) z = qf . k_sum + eps
  ksum_kernel<<<dim3(SD_/256, NB_), blk, 0, stream>>>(kfeat, ksum);
  z_kernel<<<dim3((MT * 32) / NT), blk, 0, stream>>>(qfeat, ksum, zbuf);

  // 7) kv[b] = kf[b]^T @ v[b]   split-K=4, f32 atomic-add accumulation
  hipMemsetAsync(kv, 0, (size_t)NB_*SD_*DM_*4, stream);
  gau_gemm<false,true,false,true,false,EPI_NONE,false,true>
    <<<dim3(DM_/BN, SD_/BM, NB_*KSPLIT), blk, 0, stream>>>(
      kfeat, SD_, (long)TK_*SD_, vfeat, DM_, (long)TK_*DM_, nullptr,
      kv, DM_, (long)SD_*DM_,
      nullptr, 0, nullptr, nullptr, 0, SD_, DM_, TK_, KSPLIT);

  // 8) ua[b] = u * (qf[b] @ kv[b]) / z                bf16 out, fused GAU epilogue
  gau_gemm<false,false,true,true,true,EPI_GAU,false,false>
    <<<dim3(DM_/BN, TQ_/BM, NB_), blk, 0, stream>>>(
      qfeat, SD_, (long)TQ_*SD_, kv, DM_, (long)SD_*DM_, nullptr,
      ua, DM_, (long)TQ_*DM_,
      nullptr, 0, zbuf, u, (long)TQ_*DM_, TQ_, DM_, SD_, 1);

  // 9) out = (ua Wo^T + bo) * query_mask              f32 out -> d_out
  gau_gemm<false,false,true,false,false,EPI_NONE,true,false>
    <<<dim3(DM_/BN, MT/BM, 1), blk, 0, stream>>>(
      ua, DM_, 0, Wo, DM_, 0, bo, (float*)d_out, DM_, 0,
      qmask, 0, nullptr, nullptr, 0, MT, DM_, DM_, 1);
}
